// LSTMTagger_60284160967387
// MI455X (gfx1250) — compile-verified, matrix operand order checked by reading, not run on hardware
//
#include <hip/hip_runtime.h>
#include <hip/hip_bf16.h>
#include <math.h>

// ---------------- problem constants (from reference) ----------------
#define NW    16384   // N_WORDS
#define MC    16      // MAX_CHARS
#define CE    32      // char embedding
#define H1    64      // char LSTM hidden
#define FH1   256     // 4*H1
#define WE    64      // word embedding
#define H2    128     // word LSTM hidden
#define FH2   512     // 4*H2
#define CVOC  128     // CHAR_VOCAB
#define NTAGS 64

typedef __attribute__((ext_vector_type(16))) _Float16 v16h;
typedef __attribute__((ext_vector_type(8)))  _Float16 v8h;
typedef __attribute__((ext_vector_type(8)))  float    v8f;

// D = A(16x32 f16) * B(32x16 f16) + C(16x16 f32)
__device__ __forceinline__ v8f wmma_f16(v16h a, v16h b, v8f c) {
  return __builtin_amdgcn_wmma_f32_16x16x32_f16(
      /*neg_a=*/false, a, /*neg_b=*/false, b,
      /*c_mod=*/(short)0, c, /*reuse_a=*/false, /*reuse_b=*/false);
}

// Build a 16-half fragment from two contiguous 8-half (16B) chunks.
__device__ __forceinline__ v16h ld16h(const _Float16* lo, const _Float16* hi) {
  v8h a = *(const v8h*)lo;
  v8h b = *(const v8h*)hi;
  v16h r;
#pragma unroll
  for (int i = 0; i < 8; ++i) { r[i] = a[i]; r[8 + i] = b[i]; }
  return r;
}

// Same, but converting from f32 source (weights kept f32 in HBM/L2).
__device__ __forceinline__ v16h ld16f(const float* lo, const float* hi) {
  v16h r;
#pragma unroll
  for (int i = 0; i < 8; ++i) {
    r[i]     = (_Float16)lo[i];
    r[8 + i] = (_Float16)hi[i];
  }
  return r;
}

__device__ __forceinline__ float sigf(float x) { return 1.0f / (1.0f + __expf(-x)); }

// =====================================================================
// Kernel 1: char embedding gather + batched char LSTM (16 steps).
// 2 waves / block, each wave owns a 16-word M-tile. 48 WMMAs per step.
// Writes wx = concat(word_emb[word_ids], h_char) as f16 [NW][128].
// =====================================================================
__global__ __launch_bounds__(64) void char_lstm_kernel(
    const int* __restrict__ char_ids, const int* __restrict__ word_ids,
    const int* __restrict__ char_lens,
    const float* __restrict__ char_emb, const float* __restrict__ word_emb,
    const float* __restrict__ Wih1, const float* __restrict__ Whh1,
    const float* __restrict__ bih1, const float* __restrict__ bhh1,
    _Float16* __restrict__ wx)
{
  __shared__ __align__(16) _Float16 sWih1[FH1][CE];   // 16 KB
  __shared__ __align__(16) _Float16 sWhh1[FH1][H1];   // 32 KB
  __shared__ __align__(16) _Float16 sEmb[CVOC][CE];   //  8 KB
  __shared__ __align__(16) _Float16 sH[2][16][H1];    //  4 KB
  __shared__ float sB1[FH1];                          //  1 KB

  const int tid   = threadIdx.x;
  const int wave  = tid >> 5;
  const int lane  = tid & 31;
  const int lhalf = lane & 15;
  const int hi    = lane >> 4;     // 0: K 0-7,16-23 / rows 0-7 ; 1: K 8-15,24-31 / rows 8-15
  const int k0    = hi * 8;

  for (int i = tid; i < FH1 * CE; i += 64) sWih1[i >> 5][i & 31] = (_Float16)Wih1[i];
  for (int i = tid; i < FH1 * H1; i += 64) sWhh1[i >> 6][i & 63] = (_Float16)Whh1[i];
  for (int i = tid; i < CVOC * CE; i += 64) sEmb[i >> 5][i & 31] = (_Float16)char_emb[i];
  for (int i = tid; i < FH1; i += 64) sB1[i] = bih1[i] + bhh1[i];
  for (int i = lane; i < 16 * H1; i += 32) ((_Float16*)sH[wave])[i] = (_Float16)0.0f;
  __syncthreads();

  const int wbase = (blockIdx.x * 2 + wave) * 16;

  // char_lens for the 8 rows this lane covers in the D-fragment layout
  int len8[8];
#pragma unroll
  for (int r = 0; r < 8; ++r) len8[r] = char_lens[wbase + r + hi * 8];

  v8f cst[4];
#pragma unroll
  for (int jc = 0; jc < 4; ++jc) cst[jc] = (v8f)(0.0f);

  for (int t = 0; t < MC; ++t) {
    // A fragments: x (char embedding of this step) and h (K=0..31, 32..63)
    const int w   = wbase + lhalf;
    const int cid = char_ids[w * MC + t];
    v16h ax  = ld16h(&sEmb[cid][k0],          &sEmb[cid][k0 + 16]);
    v16h ah0 = ld16h(&sH[wave][lhalf][k0],    &sH[wave][lhalf][k0 + 16]);
    v16h ah1 = ld16h(&sH[wave][lhalf][32+k0], &sH[wave][lhalf][32 + k0 + 16]);

    v8f acc[16];
#pragma unroll
    for (int j = 0; j < 16; ++j) {
      const int col = j * 16 + lhalf;          // output column (0..255)
      const float b = sB1[col];
      v8f d = {b, b, b, b, b, b, b, b};
      v16h bx  = ld16h(&sWih1[col][k0],        &sWih1[col][k0 + 16]);
      v16h bh0 = ld16h(&sWhh1[col][k0],        &sWhh1[col][k0 + 16]);
      v16h bh1 = ld16h(&sWhh1[col][32 + k0],   &sWhh1[col][32 + k0 + 16]);
      d = wmma_f16(ax,  bx,  d);
      d = wmma_f16(ah0, bh0, d);
      d = wmma_f16(ah1, bh1, d);
      acc[j] = d;
    }

    // LSTM cell: i = tiles 0..3, f = 4..7, g = 8..11, o = 12..15 (same lane/VGPR pos)
#pragma unroll
    for (int jc = 0; jc < 4; ++jc) {
#pragma unroll
      for (int r = 0; r < 8; ++r) {
        float ig = sigf(acc[jc][r]);
        float fg = sigf(acc[jc + 4][r]);
        float gg = tanhf(acc[jc + 8][r]);
        float og = sigf(acc[jc + 12][r]);
        float cn = fg * cst[jc][r] + ig * gg;
        float hn = og * tanhf(cn);
        bool valid = (t < len8[r]);
        if (valid) {
          cst[jc][r] = cn;
          sH[wave][r + hi * 8][jc * 16 + lhalf] = (_Float16)hn;  // masked h update
        }
      }
    }
    __syncthreads();   // make new h visible cross-lane before next step's A reads
  }

  // wx[w] = [ word_emb[word_ids[w]] (64) | h_char (64) ]  as f16
  for (int i = lane; i < 16 * WE; i += 32) {
    const int m = i / WE, q = i % WE;
    const int w = wbase + m;
    wx[w * (WE + H1) + q]      = (_Float16)word_emb[word_ids[w] * WE + q];
    wx[w * (WE + H1) + WE + q] = sH[wave][m][q];
  }
}

// =====================================================================
// Kernel 2: pre2 = wx @ Wih2^T + (bih2+bhh2)   [NW][512] f32
// 4 waves / block, 16-word tile per wave, 32 N-tiles x 4 K-tiles WMMAs.
// =====================================================================
__global__ __launch_bounds__(128) void pre2_kernel(
    const _Float16* __restrict__ wx, const float* __restrict__ Wih2,
    const float* __restrict__ bih2, const float* __restrict__ bhh2,
    float* __restrict__ pre2)
{
  const int tid   = threadIdx.x;
  const int wave  = tid >> 5;
  const int lane  = tid & 31;
  const int lhalf = lane & 15;
  const int hi    = lane >> 4;
  const int k0    = hi * 8;
  const int wbase = (blockIdx.x * 4 + wave) * 16;
  const int w     = wbase + lhalf;

  // A fragments for K = 0..127 (4 chunks of 32), loaded once from global f16
  v16h a[4];
#pragma unroll
  for (int kt = 0; kt < 4; ++kt)
    a[kt] = ld16h(&wx[w * H2 + kt * 32 + k0], &wx[w * H2 + kt * 32 + k0 + 16]);

  for (int j = 0; j < 32; ++j) {
    const int col = j * 16 + lhalf;          // 0..511
    const float b = bih2[col] + bhh2[col];
    v8f d = {b, b, b, b, b, b, b, b};
#pragma unroll
    for (int kt = 0; kt < 4; ++kt) {
      const float* row = &Wih2[col * H2 + kt * 32];
      v16h bw = ld16f(row + k0, row + k0 + 16);
      d = wmma_f16(a[kt], bw, d);
    }
#pragma unroll
    for (int r = 0; r < 8; ++r)
      pre2[(size_t)(wbase + r + hi * 8) * FH2 + col] = d[r];
  }
}

// =====================================================================
// Kernel 3: sequential word-LSTM scan, WMMA-based recurrent matvec.
// One 512-thread workgroup = 16 waves; wave wv owns N-tiles 2*wv, 2*wv+1.
// Whh2 fragments (8 x v16h = 64 VGPRs) pinned in registers, loaded once.
// Only D-row 0 is meaningful (M=1 matvec); pre2[t] is register-double-
// buffered so the recurrent critical path never waits on global memory.
// =====================================================================
__global__ __launch_bounds__(512) void word_lstm_kernel(
    const float* __restrict__ pre2, const float* __restrict__ Whh2,
    float* __restrict__ hs)
{
  __shared__ __align__(16) _Float16 hsh[H2];   // recurrent h, f16
  __shared__ float pre_sh[FH2];

  const int tid   = threadIdx.x;     // 0..511
  const int wv    = tid >> 5;        // 0..15
  const int lane  = tid & 31;
  const int lhalf = lane & 15;
  const int hi    = lane >> 4;
  const int k0    = hi * 8;

  // Loop-invariant B fragments for this wave's two N-tiles of Whh2^T.
  v16h bfrag[2][4];
  int col[2];
#pragma unroll
  for (int tl = 0; tl < 2; ++tl) {
    col[tl] = (2 * wv + tl) * 16 + lhalf;     // 0..511
#pragma unroll
    for (int kt = 0; kt < 4; ++kt) {
      const float* row = &Whh2[col[tl] * H2 + kt * 32];
      bfrag[tl][kt] = ld16f(row + k0, row + k0 + 16);
    }
  }

  float cst = 0.0f;                  // cell state, threads 0..127
  if (tid < H2) hsh[tid] = (_Float16)0.0f;

  // prime the pre2 register double-buffer for t = 0
  float pc0 = pre2[col[0]];
  float pc1 = pre2[col[1]];
  __syncthreads();

  for (int t = 0; t < NW; ++t) {
    // prefetch next step's input projection (independent of h)
    const int tn = (t + 1 < NW) ? (t + 1) : (NW - 1);
    float pn0 = pre2[(size_t)tn * FH2 + col[0]];
    float pn1 = pre2[(size_t)tn * FH2 + col[1]];
    __builtin_prefetch(&pre2[(size_t)((t + 16 < NW) ? t + 16 : NW - 1) * FH2 + col[0]], 0, 1);

    // A fragments: row 0 = h, rows 1..15 = 0 (only lanes 0/16 carry row 0)
    v16h a[4];
#pragma unroll
    for (int kt = 0; kt < 4; ++kt) a[kt] = (v16h)(_Float16)0.0f;
    if (lhalf == 0) {
#pragma unroll
      for (int kt = 0; kt < 4; ++kt)
        a[kt] = ld16h(&hsh[kt * 32 + k0], &hsh[kt * 32 + k0 + 16]);
    }

    // two N-tiles: d = h @ Whh2^T + pre2[t]
#pragma unroll
    for (int tl = 0; tl < 2; ++tl) {
      const float p = (tl == 0) ? pc0 : pc1;
      v8f d = {p, p, p, p, p, p, p, p};
#pragma unroll
      for (int kt = 0; kt < 4; ++kt) d = wmma_f16(a[kt], bfrag[tl][kt], d);
      if (lane < 16) pre_sh[col[tl]] = d[0];   // D row 0: VGPR0, lanes 0..15
    }
    __syncthreads();

    if (tid < H2) {
      float ig = sigf(pre_sh[tid]);
      float fg = sigf(pre_sh[H2 + tid]);
      float gg = tanhf(pre_sh[2 * H2 + tid]);
      float og = sigf(pre_sh[3 * H2 + tid]);
      cst = fg * cst + ig * gg;
      float hn = og * tanhf(cst);
      hsh[tid] = (_Float16)hn;
      hs[(size_t)t * H2 + tid] = hn;           // fire-and-forget store
    }
    __syncthreads();

    pc0 = pn0;
    pc1 = pn1;
  }
}

// =====================================================================
// Kernel 4: logits = hs @ Wtag^T + btag, then row log_softmax (64 tags).
// One wave / block, 16-word tile, 4 N-tiles x 4 K-tiles WMMAs.
// =====================================================================
__global__ __launch_bounds__(32) void tag_kernel(
    const float* __restrict__ hs, const float* __restrict__ Wtag,
    const float* __restrict__ btag, float* __restrict__ out)
{
  __shared__ __align__(16) float sLog[16][NTAGS];   // 4 KB
  const int lane  = threadIdx.x;
  const int lhalf = lane & 15;
  const int hi    = lane >> 4;
  const int k0    = hi * 8;
  const int wbase = blockIdx.x * 16;
  const int w     = wbase + lhalf;

  v16h a[4];
#pragma unroll
  for (int kt = 0; kt < 4; ++kt) {
    const float* row = &hs[(size_t)w * H2 + kt * 32];
    a[kt] = ld16f(row + k0, row + k0 + 16);
  }

#pragma unroll
  for (int j = 0; j < 4; ++j) {
    const int col = j * 16 + lhalf;          // 0..63
    const float b = btag[col];
    v8f d = {b, b, b, b, b, b, b, b};
#pragma unroll
    for (int kt = 0; kt < 4; ++kt) {
      const float* row = &Wtag[col * H2 + kt * 32];
      v16h bw = ld16f(row + k0, row + k0 + 16);
      d = wmma_f16(a[kt], bw, d);
    }
#pragma unroll
    for (int r = 0; r < 8; ++r) sLog[r + hi * 8][col] = d[r];
  }
  __syncthreads();

  if (lane < 16) {
    const int m = lane;
    float mx = -1e30f;
    for (int n = 0; n < NTAGS; ++n) mx = fmaxf(mx, sLog[m][n]);
    float s = 0.0f;
    for (int n = 0; n < NTAGS; ++n) s += __expf(sLog[m][n] - mx);
    const float lse = mx + __logf(s);
    for (int n = 0; n < NTAGS; ++n)
      out[(size_t)(wbase + m) * NTAGS + n] = sLog[m][n] - lse;
  }
}

// =====================================================================
extern "C" void kernel_launch(void* const* d_in, const int* in_sizes, int n_in,
                              void* d_out, int out_size, void* d_ws, size_t ws_size,
                              hipStream_t stream) {
  const int*   char_ids  = (const int*)d_in[0];
  const int*   word_ids  = (const int*)d_in[1];
  const int*   char_lens = (const int*)d_in[2];
  const float* char_emb  = (const float*)d_in[3];
  const float* word_emb  = (const float*)d_in[4];
  const float* Wih1      = (const float*)d_in[5];
  const float* Whh1      = (const float*)d_in[6];
  const float* bih1      = (const float*)d_in[7];
  const float* bhh1      = (const float*)d_in[8];
  const float* Wih2      = (const float*)d_in[9];
  const float* Whh2      = (const float*)d_in[10];
  const float* bih2      = (const float*)d_in[11];
  const float* bhh2      = (const float*)d_in[12];
  const float* Wtag      = (const float*)d_in[13];
  const float* btag      = (const float*)d_in[14];

  // workspace layout: wx f16 (4 MB) | pre2 f32 (32 MB) | hs f32 (8 MB)
  char* ws = (char*)d_ws;
  _Float16* wx  = (_Float16*)ws;
  float*    pre2 = (float*)(ws + (size_t)NW * H2 * sizeof(_Float16));
  float*    hs   = (float*)(ws + (size_t)NW * H2 * sizeof(_Float16)
                               + (size_t)NW * FH2 * sizeof(float));

  char_lstm_kernel<<<NW / 32, 64, 0, stream>>>(
      char_ids, word_ids, char_lens, char_emb, word_emb,
      Wih1, Whh1, bih1, bhh1, wx);

  pre2_kernel<<<NW / 64, 128, 0, stream>>>(wx, Wih2, bih2, bhh2, pre2);

  word_lstm_kernel<<<1, 512, 0, stream>>>(pre2, Whh2, hs);

  tag_kernel<<<NW / 16, 32, 0, stream>>>(hs, Wtag, btag, (float*)d_out);
}
